// TextClassification_26740466385737
// MI455X (gfx1250) — compile-verified, matrix operand order checked by reading, not run on hardware
//
#include <hip/hip_runtime.h>

// ---------------------------------------------------------------------------
// CDNA5 (gfx1250) LSTM text classifier: bf16 WMMA path.
//   V=50257, D=256, H=256, O=10, L=2048, N=64
// ---------------------------------------------------------------------------

typedef __attribute__((ext_vector_type(16))) __bf16 v16bf;
typedef __attribute__((ext_vector_type(8)))  __bf16 v8bf;
typedef __attribute__((ext_vector_type(8)))  float  v8f;

#define L_SEQ 2048
#define NB    64
#define DIM   256
#define HID   256
#define OUTD  10
#define G4    1024   // 4 * HID (gate order: i, f, g, o)
// G buffer is fragment-major: [step][tile=rt*64+ctF][lane][v], 8 f32 per lane.
#define G_STEP (NB * G4)          // 65536 f32 per step
#define G_TILE (32 * 8)           // 256 f32 per 16x16 tile

__device__ __forceinline__ float sigf(float x) { return 1.0f / (1.0f + __expf(-x)); }

__device__ __forceinline__ v16bf cat8(v8bf a, v8bf b) {
  return __builtin_shufflevector(a, b, 0,1,2,3,4,5,6,7,8,9,10,11,12,13,14,15);
}

// A/B fragment for V_WMMA_F32_16X16X32_BF16 (ISA 7.12.2 16-bit 16x32 layout):
// lane&15 selects the row; lanes 0-15 hold K = {kwin+0..7, kwin+16..23},
// lanes 16-31 hold K = {kwin+8..15, kwin+24..31}. Two contiguous 8-elem runs.
__device__ __forceinline__ v16bf frag_from_bf16(const __bf16* row, int kwin, int hi) {
  v8bf b1 = *reinterpret_cast<const v8bf*>(row + kwin + hi * 8);
  v8bf b2 = *reinterpret_cast<const v8bf*>(row + kwin + 16 + hi * 8);
  return cat8(b1, b2);
}

__device__ __forceinline__ v8f wmma_bf16(v16bf a, v16bf b, v8f c) {
  // (neg_a, A, neg_b, B, c_mod, C, reuse_a, reuse_b)
  return __builtin_amdgcn_wmma_f32_16x16x32_bf16(false, a, false, b, (short)0, c, false, false);
}

// ---------------------------------------------------------------------------
// Kernel 1: pack weights to bf16. grid = 1024 blocks (one per gate-unit row c),
// block = 256 threads (one per k). Also builds combined bias b_in + b_hid.
// ---------------------------------------------------------------------------
__global__ void lstm_pack_kernel(
    const float* Wii, const float* Wif, const float* Wig, const float* Wio,
    const float* Whi, const float* Whf, const float* Whg, const float* Who,
    const float* bii, const float* bif, const float* big, const float* bio,
    const float* bhi, const float* bhf, const float* bhg, const float* bho,
    __bf16* Wx, __bf16* Wh, float* bias) {
  int c = blockIdx.x;          // 0..1023
  int d = threadIdx.x;         // 0..255
  int gate = c >> 8;           // 0:i 1:f 2:g 3:o
  int u = c & 255;
  const float *wi, *wh, *bi, *bh;
  if (gate == 0)      { wi = Wii; wh = Whi; bi = bii; bh = bhi; }
  else if (gate == 1) { wi = Wif; wh = Whf; bi = bif; bh = bhf; }
  else if (gate == 2) { wi = Wig; wh = Whg; bi = big; bh = bhg; }
  else                { wi = Wio; wh = Who; bi = bio; bh = bho; }
  Wx[c * DIM + d] = (__bf16)wi[u * DIM + d];
  Wh[c * HID + d] = (__bf16)wh[u * HID + d];
  if (d == 0) bias[c] = bi[u] + bh[u];
}

// ---------------------------------------------------------------------------
// Kernel 2: zero initial h (bf16) and c (f32) state.
// ---------------------------------------------------------------------------
__global__ void lstm_init_kernel(float* c_ws, unsigned* h_ws) {
  int i = blockIdx.x * blockDim.x + threadIdx.x;
  if (i < NB * HID)          c_ws[i] = 0.0f;           // 16384 floats
  if (i < (NB * HID) / 2)    h_ws[i] = 0u;             // 8192 dwords of bf16x2
}

// ---------------------------------------------------------------------------
// Kernel 3: input-side gate projections, one block per timestep (8 waves).
// Phase 1: gather xe_t = emb[x[t]] into LDS once, converting f32->bf16.
// Phase 2: wave w computes col tiles ctF = w*8..w*8+7 for all 4 row tiles;
//          each B fragment feeds 4 WMMAs (independent accumulator chains).
// Output G is fragment-major so the serial kernel reloads it with v8f loads.
// ---------------------------------------------------------------------------
__global__ void __launch_bounds__(256)
lstm_proj_kernel(const int* __restrict__ x, const float* __restrict__ emb,
                 const __bf16* __restrict__ Wx, const float* __restrict__ bias,
                 float* __restrict__ G, int l0) {
  __shared__ __attribute__((aligned(16))) __bf16 xsh[NB * DIM];   // 32 KB

  int tid  = threadIdx.x;
  int w    = tid >> 5;
  int lane = tid & 31;
  int l15  = lane & 15;
  int hi   = lane >> 4;
  int s    = blockIdx.x;
  int l    = l0 + s;

  // Phase 1: embedding gather + bf16 convert (4 threads per batch row)
  {
    int n  = tid >> 2;                 // 0..63
    int c0 = (tid & 3) * 64;           // 64 contiguous d's per thread
    int token = x[l * NB + n];
    const float* src = emb + (long)token * DIM + c0;
    __bf16* dst = xsh + n * DIM + c0;
#pragma unroll
    for (int i = 0; i < 8; ++i) {
      v8f f = *reinterpret_cast<const v8f*>(src + i * 8);
      *reinterpret_cast<v8bf*>(dst + i * 8) = __builtin_convertvector(f, v8bf);
    }
  }
  __syncthreads();

  float* gs = G + (long)s * G_STEP;

  for (int j = 0; j < 8; ++j) {
    int ctF = w * 8 + j;
    int col = ctF * 16 + l15;
    const __bf16* brow = Wx + (long)col * DIM;
    float bv = bias[col];
    v8f acc[4];
#pragma unroll
    for (int rt = 0; rt < 4; ++rt) acc[rt] = {bv, bv, bv, bv, bv, bv, bv, bv};

#pragma unroll
    for (int ki = 0; ki < 8; ++ki) {               // K = 256 in steps of 32
      v16bf b = frag_from_bf16(brow, ki * 32, hi); // one B feeds 4 row tiles
#pragma unroll
      for (int rt = 0; rt < 4; ++rt) {
        v16bf a = frag_from_bf16(xsh + (rt * 16 + l15) * DIM, ki * 32, hi);
        acc[rt] = wmma_bf16(a, b, acc[rt]);
      }
    }

#pragma unroll
    for (int rt = 0; rt < 4; ++rt) {
      int tile = rt * 64 + ctF;
      *reinterpret_cast<v8f*>(gs + (long)tile * G_TILE + lane * 8) = acc[rt];
    }
  }
}

// ---------------------------------------------------------------------------
// Kernel 4: serial recurrence. One persistent 1024-thread workgroup (32 waves).
// h (bf16) lives in 32 KB LDS; c lives in registers (16 cells/thread).
// Wave w owns groups {2w, 2w+1}: group g -> row tile rt=g>>4, h-col tile g&15;
// each group accumulates all 4 gate tiles so the nonlinearity fuses in-reg.
// ---------------------------------------------------------------------------
__global__ void __launch_bounds__(1024)
lstm_rec_kernel(const float* __restrict__ G, const __bf16* __restrict__ Wh,
                float* __restrict__ c_ws, unsigned* __restrict__ h_ws, int steps) {
  __shared__ __attribute__((aligned(16))) __bf16 hsh[NB * HID];   // 32 KB

  int tid  = threadIdx.x;
  int w    = tid >> 5;
  int lane = tid & 31;
  int l15  = lane & 15;
  int hi   = lane >> 4;

  // restore h state into LDS (16 bf16 = 8 dwords per thread)
  unsigned* hs32 = reinterpret_cast<unsigned*>(hsh);
#pragma unroll
  for (int i = 0; i < 8; ++i) hs32[tid * 8 + i] = h_ws[tid * 8 + i];

  // restore c state into registers
  float creg[2][8];
#pragma unroll
  for (int gi = 0; gi < 2; ++gi)
#pragma unroll
    for (int v = 0; v < 8; ++v)
      creg[gi][v] = c_ws[((w * 2 + gi) * 8 + v) * 32 + lane];

  __syncthreads();

  int grp0 = w * 2;
  int rt   = grp0 >> 4;                       // shared by both groups of a wave
  const __bf16* arow = hsh + (rt * 16 + l15) * HID;

  for (int s = 0; s < steps; ++s) {
    const float* gstep = G + (long)s * G_STEP;
    if (s + 1 < steps)                        // hint next step's gates into cache
      __builtin_prefetch(gstep + G_STEP + tid * 64, 0, 0);

    // init accumulators with precomputed input-side pre-activations:
    // fragment-major layout -> one coalesced v8f load per accumulator
    v8f acc[2][4];
#pragma unroll
    for (int gi = 0; gi < 2; ++gi) {
      int ctH = (grp0 + gi) & 15;
#pragma unroll
      for (int gate = 0; gate < 4; ++gate) {
        int tile = rt * 64 + gate * 16 + ctH;
        acc[gi][gate] =
            *reinterpret_cast<const v8f*>(gstep + (long)tile * G_TILE + lane * 8);
      }
    }

    // S += h @ Wh^T : one shared A-fragment feeds 8 accumulators per K step
#pragma unroll
    for (int ki = 0; ki < 8; ++ki) {
      v16bf a = frag_from_bf16(arow, ki * 32, hi);      // ds_load from LDS
#pragma unroll
      for (int gi = 0; gi < 2; ++gi) {
        int ctH = (grp0 + gi) & 15;
#pragma unroll
        for (int gate = 0; gate < 4; ++gate) {
          const __bf16* brow = Wh + (long)(gate * HID + ctH * 16 + l15) * HID;
          v16bf b = frag_from_bf16(brow, ki * 32, hi);  // L2-resident (512 KB)
          acc[gi][gate] = wmma_bf16(a, b, acc[gi][gate]);
        }
      }
    }

    // fused LSTM cell nonlinearity; c stays in registers
    __bf16 hout[2][8];
#pragma unroll
    for (int gi = 0; gi < 2; ++gi) {
#pragma unroll
      for (int v = 0; v < 8; ++v) {
        float iv = sigf(acc[gi][0][v]);
        float fv = sigf(acc[gi][1][v]);
        float gv = tanhf(acc[gi][2][v]);
        float ov = sigf(acc[gi][3][v]);
        float c  = fv * creg[gi][v] + iv * gv;
        creg[gi][v] = c;
        hout[gi][v] = (__bf16)(ov * tanhf(c));
      }
    }

    __syncthreads();   // all waves finished reading old h
#pragma unroll
    for (int gi = 0; gi < 2; ++gi) {
      int ctH = (grp0 + gi) & 15;
      int col = ctH * 16 + l15;
#pragma unroll
      for (int v = 0; v < 8; ++v) {
        int n = rt * 16 + v + 8 * hi;
        hsh[n * HID + col] = hout[gi][v];
      }
    }
    __syncthreads();   // new h visible to everyone
  }

  // persist state for the next chunk launch
#pragma unroll
  for (int gi = 0; gi < 2; ++gi)
#pragma unroll
    for (int v = 0; v < 8; ++v)
      c_ws[((w * 2 + gi) * 8 + v) * 32 + lane] = creg[gi][v];
#pragma unroll
  for (int i = 0; i < 8; ++i) h_ws[tid * 8 + i] = hs32[tid * 8 + i];
}

// ---------------------------------------------------------------------------
// Kernel 5: output head: out[n][o] = h[n] . Wout[o] + bout[o]  (64x10, tiny)
// ---------------------------------------------------------------------------
__global__ void lstm_head_kernel(const __bf16* __restrict__ h,
                                 const float* __restrict__ Wout,
                                 const float* __restrict__ bout,
                                 float* __restrict__ out) {
  int tid = threadIdx.x;               // 640 threads
  if (tid >= NB * OUTD) return;
  int n = tid / OUTD, o = tid % OUTD;
  float acc = bout[o];
  for (int k = 0; k < HID; ++k)
    acc += (float)h[n * HID + k] * Wout[o * HID + k];
  out[n * OUTD + o] = acc;
}

// ---------------------------------------------------------------------------
// Host launcher
// ---------------------------------------------------------------------------
extern "C" void kernel_launch(void* const* d_in, const int* in_sizes, int n_in,
                              void* d_out, int out_size, void* d_ws, size_t ws_size,
                              hipStream_t stream) {
  (void)in_sizes; (void)n_in; (void)out_size;
  const int*   x    = (const int*)  d_in[0];
  const float* emb  = (const float*)d_in[1];
  const float* Wii  = (const float*)d_in[2];   const float* bii = (const float*)d_in[3];
  const float* Whi  = (const float*)d_in[4];   const float* bhi = (const float*)d_in[5];
  const float* Wif  = (const float*)d_in[6];   const float* bif = (const float*)d_in[7];
  const float* Whf  = (const float*)d_in[8];   const float* bhf = (const float*)d_in[9];
  const float* Wig  = (const float*)d_in[10];  const float* big = (const float*)d_in[11];
  const float* Whg  = (const float*)d_in[12];  const float* bhg = (const float*)d_in[13];
  const float* Wio  = (const float*)d_in[14];  const float* bio = (const float*)d_in[15];
  const float* Who  = (const float*)d_in[16];  const float* bho = (const float*)d_in[17];
  const float* Wout = (const float*)d_in[18];  const float* bout= (const float*)d_in[19];

  // workspace layout
  char* ws = (char*)d_ws;
  __bf16*   Wx   = (__bf16*)(ws);                                    // 512 KB
  __bf16*   Wh   = (__bf16*)(ws + 512 * 1024);                       // 512 KB
  float*    bias = (float*) (ws + 1024 * 1024);                      //   4 KB
  float*    c_ws = (float*) (ws + 1024 * 1024 + 4096);               //  64 KB
  unsigned* h_ws = (unsigned*)(ws + 1024 * 1024 + 4096 + 65536);     //  32 KB
  float*    G    = (float*) (ws + 2u * 1024 * 1024);                 // CH*256 KB

  // adapt chunk length to available workspace (gate buffer = 256 KB / step)
  const size_t fixed = 2u * 1024 * 1024;
  const size_t per_step = (size_t)G_STEP * sizeof(float);
  long chunk = 256;
  if (ws_size > fixed + per_step) {
    long avail = (long)((ws_size - fixed) / per_step);
    if (avail < chunk) chunk = avail;
  } else {
    chunk = 1;
  }
  if (chunk > L_SEQ) chunk = L_SEQ;
  if (chunk < 1) chunk = 1;

  lstm_pack_kernel<<<dim3(G4), dim3(DIM), 0, stream>>>(
      Wii, Wif, Wig, Wio, Whi, Whf, Whg, Who,
      bii, bif, big, bio, bhi, bhf, bhg, bho, Wx, Wh, bias);

  lstm_init_kernel<<<dim3((NB * HID + 255) / 256), dim3(256), 0, stream>>>(c_ws, h_ws);

  for (int l0 = 0; l0 < L_SEQ; l0 += (int)chunk) {
    int steps = L_SEQ - l0;
    if (steps > (int)chunk) steps = (int)chunk;
    lstm_proj_kernel<<<dim3(steps), dim3(256), 0, stream>>>(x, emb, Wx, bias, G, l0);
    lstm_rec_kernel<<<dim3(1), dim3(1024), 0, stream>>>(G, Wh, c_ws, h_ws, steps);
  }

  lstm_head_kernel<<<dim3(1), dim3(640), 0, stream>>>((const __bf16*)h_ws, Wout, bout,
                                                      (float*)d_out);
}
